// TwoWordNonPSDProbe_4148938408316
// MI455X (gfx1250) — compile-verified, matrix operand order checked by reading, not run on hardware
//
#include <hip/hip_runtime.h>

// ---------------------------------------------------------------------------
// dists[b,i,j] = q_i + q_j - cross_ij - cross_ji, cross = (H A) H^T
// With M = A + A^T (symmetric): sym = H M H^T (symmetric),
//   dists[i,j] = 0.5*sym_ii + 0.5*sym_jj - sym_ij
// Pipeline: M_h=f16(A+A^T) ; H_h=f16(H) ; P_h = H_h*M_h (WMMA f16->f32->f16) ;
//           q = 0.5*rowdot(P_h,H_h) ; dists from triangular 64x64 square tiles
//           (2 waves per tile, each 64x32). 64x32 per-wave tiles: 8 WMMA per
//           12 b128 fragment loads, ~180 VGPRs -> no spills w/ launch_bounds.
// ---------------------------------------------------------------------------

typedef _Float16 f16;
typedef _Float16 v4h  __attribute__((ext_vector_type(4)));
typedef _Float16 v8h  __attribute__((ext_vector_type(8)));
typedef _Float16 v16h __attribute__((ext_vector_type(16)));
typedef float    v4f  __attribute__((ext_vector_type(4)));
typedef float    v8f  __attribute__((ext_vector_type(8)));

#define D_DIM 768
#define S_DIM 1024
#define B_DIM 64

// Load a 16x32 f16 WMMA A/B fragment whose rows (A: M, B: N) are contiguous
// row-major with stride D_DIM halves. CDNA5 layout per lane:
//   row = lane%16 ; chunks of 8 halves at k = (lane/16)*8 and (lane/16)*8+16.
static __device__ __forceinline__ v16h load_frag(const f16* __restrict__ base,
                                                 int lane) {
  const int row = lane & 15;
  const int kb  = (lane >> 4) << 3;
  const f16* p  = base + (size_t)row * D_DIM + kb;
  v8h lo = *(const v8h*)(p);
  v8h hi = *(const v8h*)(p + 16);
  return __builtin_shufflevector(lo, hi, 0, 1, 2, 3, 4, 5, 6, 7,
                                 8, 9, 10, 11, 12, 13, 14, 15);
}

#define WMMA_F16(A, B, C) \
  __builtin_amdgcn_wmma_f32_16x16x32_f16(false, (A), false, (B), (short)0, \
                                         (C), false, false)

// --------------------------- kernel 1: M = A + A^T -> f16 -------------------
__global__ __launch_bounds__(256) void msym_kernel(
    const float* __restrict__ proj, f16* __restrict__ msym) {
  int idx = blockIdx.x * blockDim.x + threadIdx.x;     // over D*D
  int d = idx / D_DIM;
  int e = idx - d * D_DIM;
  msym[idx] = (f16)(proj[idx] + proj[e * D_DIM + d]);
}

// --------------------------- kernel 2: f32 -> f16 (x4) ----------------------
__global__ __launch_bounds__(256) void cvt_kernel(const float* __restrict__ x,
                                                  f16* __restrict__ xh) {
  size_t i = (size_t)blockIdx.x * blockDim.x + threadIdx.x;  // over n/4
  v4f v = *(const v4f*)(x + i * 4);
  v4h h;
  h.x = (f16)v.x; h.y = (f16)v.y; h.z = (f16)v.z; h.w = (f16)v.w;
  *(v4h*)(xh + i * 4) = h;
}

// ------------------- kernel 3: P_h = H_h * M_h  (WMMA) ----------------------
// Each wave owns a 64x32 output tile. B columns = rows of symmetric M.
__global__ __launch_bounds__(256) void gemm_p_kernel(
    const f16* __restrict__ hh,    // (B*S, D)
    const f16* __restrict__ msym,  // (D, D) symmetric
    f16* __restrict__ ph) {        // (B*S, D)
  const int lane = threadIdx.x & 31;
  const int wave = threadIdx.x >> 5;
  const int tile = blockIdx.x * 8 + wave;
  const int NT = D_DIM / 32;                 // 24 tiles along N
  const int mt = tile / NT;
  const int nt = tile - mt * NT;
  const size_t m0 = (size_t)mt * 64;
  const int n0 = nt * 32;

  v8f acc[4][2] = {};
  for (int k0 = 0; k0 < D_DIM; k0 += 32) {
    v16h a[4], b[2];
#pragma unroll
    for (int t = 0; t < 4; t++)
      a[t] = load_frag(hh + (m0 + t * 16) * D_DIM + k0, lane);
#pragma unroll
    for (int t = 0; t < 2; t++)
      b[t] = load_frag(msym + (size_t)(n0 + t * 16) * D_DIM + k0, lane);
#pragma unroll
    for (int mi = 0; mi < 4; mi++)
#pragma unroll
      for (int ni = 0; ni < 2; ni++)
        acc[mi][ni] = WMMA_F16(a[mi], b[ni], acc[mi][ni]);
  }
  // C layout: VGPR r -> M = r + 8*(lane/16), N = lane%16
  const int ncol = lane & 15;
  const int mofs = (lane >> 4) << 3;
#pragma unroll
  for (int mi = 0; mi < 4; mi++) {
#pragma unroll
    for (int ni = 0; ni < 2; ni++) {
      const int n = n0 + ni * 16 + ncol;
      const size_t mb = m0 + mi * 16 + mofs;
#pragma unroll
      for (int r = 0; r < 8; r++)
        ph[(mb + r) * D_DIM + n] = (f16)acc[mi][ni][r];
    }
  }
}

// ------------------- kernel 4: q = 0.5 * rowdot(P_h, H_h) -------------------
__global__ __launch_bounds__(256) void q_kernel(const f16* __restrict__ ph,
                                                const f16* __restrict__ hh,
                                                float* __restrict__ q) {
  const int lane = threadIdx.x & 31;
  const int wave = threadIdx.x >> 5;
  const size_t row = (size_t)blockIdx.x * 8 + wave;  // over B*S rows
  const f16* p = ph + row * D_DIM;
  const f16* h = hh + row * D_DIM;
  float s = 0.f;
#pragma unroll
  for (int c = 0; c < 3; c++) {
    v8h pv = *(const v8h*)(p + c * 256 + lane * 8);
    v8h hv = *(const v8h*)(h + c * 256 + lane * 8);
#pragma unroll
    for (int j = 0; j < 8; j++) s += (float)pv[j] * (float)hv[j];
  }
#pragma unroll
  for (int off = 16; off > 0; off >>= 1) s += __shfl_down(s, off, 32);
  if (lane == 0) q[row] = 0.5f * s;
}

// ------- kernel 5: triangular 64x64 sym tiles, 2 waves per tile -------------
// Wave g = (pair, half): computes rows i0..i0+63 x cols j0+32*half..+31.
// Mirror writes only for off-diagonal square tiles -> each output element is
// written exactly once (deterministic).
__global__ __launch_bounds__(256) void gemm_dist_kernel(
    const f16* __restrict__ ph,   // (B,S,D)
    const f16* __restrict__ hh,   // (B,S,D)
    const float* __restrict__ q,  // (B,S)
    float* __restrict__ out) {    // (B,S,S)
  const int T = S_DIM / 64;          // 16 square tiles per dim
  const int lane = threadIdx.x & 31;
  const int wave = threadIdx.x >> 5;
  const int g = blockIdx.x * 8 + wave;   // 0..271
  const int p = g >> 1;                  // 0..135 (== T*(T+1)/2 - 1)
  const int half = g & 1;
  const int b = blockIdx.y;

  // decode triangular pair index -> (ti, tj), tj >= ti
  int ti = 0, rem = p;
  while (rem >= T - ti) { rem -= (T - ti); ti++; }
  const int tj = ti + rem;
  const int i0 = ti * 64;
  const int j0 = tj * 64 + half * 32;

  const f16* A  = ph + (size_t)b * S_DIM * D_DIM;
  const f16* Bm = hh + (size_t)b * S_DIM * D_DIM;

  v8f acc[4][2] = {};
  for (int k0 = 0; k0 < D_DIM; k0 += 32) {
    v16h a[4], bf[2];
#pragma unroll
    for (int t = 0; t < 4; t++)
      a[t] = load_frag(A + (size_t)(i0 + t * 16) * D_DIM + k0, lane);
#pragma unroll
    for (int t = 0; t < 2; t++)
      bf[t] = load_frag(Bm + (size_t)(j0 + t * 16) * D_DIM + k0, lane);
#pragma unroll
    for (int mi = 0; mi < 4; mi++)
#pragma unroll
      for (int ni = 0; ni < 2; ni++)
        acc[mi][ni] = WMMA_F16(a[mi], bf[ni], acc[mi][ni]);
  }

  const float* qb = q + (size_t)b * S_DIM;
  float* ob = out + (size_t)b * S_DIM * S_DIM;
  const bool mirror = (ti != tj);  // diagonal square tiles cover both orders
  const int ncol = lane & 15;
  const int mofs = (lane >> 4) << 3;
#pragma unroll
  for (int mi = 0; mi < 4; mi++) {
#pragma unroll
    for (int ni = 0; ni < 2; ni++) {
      const int n = j0 + ni * 16 + ncol;
      const float qj = qb[n];
      const int mb = i0 + mi * 16 + mofs;
#pragma unroll
      for (int r = 0; r < 8; r++) {
        const int m = mb + r;
        const float v = qb[m] + qj - acc[mi][ni][r];
        ob[(size_t)m * S_DIM + n] = v;
        if (mirror) ob[(size_t)n * S_DIM + m] = v;
      }
    }
  }
}

// ---------------------------------------------------------------------------
extern "C" void kernel_launch(void* const* d_in, const int* in_sizes, int n_in,
                              void* d_out, int out_size, void* d_ws,
                              size_t ws_size, hipStream_t stream) {
  const float* batch = (const float*)d_in[0];  // (64,1024,768) f32
  const float* proj  = (const float*)d_in[1];  // (768,768) f32
  float* out = (float*)d_out;                  // (64,1024,1024) f32

  char* ws = (char*)d_ws;
  // workspace layout (~194.3 MB total)
  f16*   msym = (f16*)(ws);                                      // 1.13 MiB
  f16*   hh   = (f16*)(ws + ((size_t)2 << 20));                  // 96 MiB
  f16*   ph   = (f16*)(ws + ((size_t)98 << 20));                 // 96 MiB
  float* q    = (float*)(ws + ((size_t)194 << 20));              // 256 KiB

  const size_t nrow = (size_t)B_DIM * S_DIM;                     // 65536

  // 1) M = A + A^T (f16)
  msym_kernel<<<(D_DIM * D_DIM) / 256, 256, 0, stream>>>(proj, msym);
  // 2) H -> f16
  cvt_kernel<<<(nrow * D_DIM / 4) / 256, 256, 0, stream>>>(batch, hh);
  // 3) P_h = H_h * M_h : (nrow/64)*(D/32) = 1024*24 = 24576 wave-tiles
  gemm_p_kernel<<<24576 / 8, 256, 0, stream>>>(hh, msym, ph);
  // 4) q = 0.5 * rowdot(P_h, H_h)
  q_kernel<<<nrow / 8, 256, 0, stream>>>(ph, hh, q);
  // 5) triangular sym tiles: 136 pairs * 2 half-waves = 272 waves per batch
  gemm_dist_kernel<<<dim3(272 / 8, B_DIM), 256, 0, stream>>>(ph, hh, q, out);
}
